// gaussian_layer_17626545783610
// MI455X (gfx1250) — compile-verified, matrix operand order checked by reading, not run on hardware
//
#include <hip/hip_runtime.h>
#include <hip/hip_bf16.h>

typedef __attribute__((ext_vector_type(16))) _Float16 v16h;
typedef __attribute__((ext_vector_type(8)))  float    v8f;
typedef __attribute__((ext_vector_type(4)))  float    f32x4;

#define DIN   128
#define DOUTC 512
#define NT    (DOUTC / 16)   // 32 column tiles
#define NKB   (DIN / 32)     // 4 K chunks

// Workspace layout (bytes):
//   [0)                : packed f16 centers, NT*NKB*32 lanes * 16 halfs = 128 KB
//   [CPK_BYTES)        : c2[512]  (fp32 squared norms of centers)
//   [CPK_BYTES + 2KB)  : inv_s2[512]
#define CPK_HALFS  (NT * NKB * 32 * 16)
#define CPK_BYTES  (CPK_HALFS * 2)          // 131072
#define C2_OFF     (CPK_BYTES)
#define IS2_OFF    (CPK_BYTES + DOUTC * 4)

// CDNA5 16-bit A/B fragment layout (wave32), 16x32 tile of a row-major matrix:
//   lanes 0-15  : row = lane,    halfs = K {kb+0..7,  kb+16..23}
//   lanes 16-31 : row = lane-16, halfs = K {kb+8..15, kb+24..31}
__device__ __forceinline__ v16h cvt_frag_f16(const float* __restrict__ rowp,
                                             int kbase, int hi, float& sumsq) {
  const float* p = rowp + kbase + (hi ? 8 : 0);
  f32x4 a0 = *(const f32x4*)(p + 0);
  f32x4 a1 = *(const f32x4*)(p + 4);
  f32x4 b0 = *(const f32x4*)(p + 16);
  f32x4 b1 = *(const f32x4*)(p + 20);
  sumsq += a0.x*a0.x + a0.y*a0.y + a0.z*a0.z + a0.w*a0.w
         + a1.x*a1.x + a1.y*a1.y + a1.z*a1.z + a1.w*a1.w
         + b0.x*b0.x + b0.y*b0.y + b0.z*b0.z + b0.w*b0.w
         + b1.x*b1.x + b1.y*b1.y + b1.z*b1.z + b1.w*b1.w;
  v16h h;
  h[0]=(_Float16)a0.x; h[1]=(_Float16)a0.y; h[2]=(_Float16)a0.z; h[3]=(_Float16)a0.w;
  h[4]=(_Float16)a1.x; h[5]=(_Float16)a1.y; h[6]=(_Float16)a1.z; h[7]=(_Float16)a1.w;
  h[8]=(_Float16)b0.x; h[9]=(_Float16)b0.y; h[10]=(_Float16)b0.z; h[11]=(_Float16)b0.w;
  h[12]=(_Float16)b1.x; h[13]=(_Float16)b1.y; h[14]=(_Float16)b1.z; h[15]=(_Float16)b1.w;
  return h;
}

// Prologue 1: repack centers (512x128 fp32, row-major) into per-lane f16
// B-fragments.  One thread = one (tile t, k-chunk kbi, lane) triple -> 32 B.
__global__ __launch_bounds__(256)
void pack_centers_kernel(const float* __restrict__ C, _Float16* __restrict__ Cpk) {
  const int id   = blockIdx.x * 256 + threadIdx.x;   // 0 .. NT*NKB*32-1
  const int lane = id & 31;
  const int kbi  = (id >> 5) & (NKB - 1);
  const int t    = id >> 7;                          // 0..NT-1
  const int hi   = (lane >= 16) ? 1 : 0;
  const int l16  = lane & 15;
  float dummy = 0.f;
  v16h h = cvt_frag_f16(C + (size_t)(t * 16 + l16) * DIN, kbi * 32, hi, dummy);
  *(v16h*)(Cpk + (size_t)id * 16) = h;
}

// Prologue 2: c2[o] = ||centers[o]||^2 (fp32), inv_s2[o] = exp(-2*log_sigma[o]).
__global__ __launch_bounds__(256)
void center_norms_kernel(const float* __restrict__ C, const float* __restrict__ LS,
                         float* __restrict__ c2, float* __restrict__ inv_s2) {
  const int o = blockIdx.x * 256 + threadIdx.x;
  if (o >= DOUTC) return;
  const float* row = C + (size_t)o * DIN;
  float s = 0.f;
#pragma unroll 4
  for (int k = 0; k < DIN; k += 4) {
    f32x4 v = *(const f32x4*)(row + k);
    s += v.x*v.x + v.y*v.y + v.z*v.z + v.w*v.w;
  }
  c2[o] = s;
  inv_s2[o] = expf(-2.0f * LS[o]);
}

// Main: block = 8 wave32s; block tile 16 rows x 512 cols; wave tile 16x64
// (4x 16x16 f32 accumulators).  B fragments are pure b128 loads from the
// packed f16 centers; A is converted on the fly (x2 falls out for free).
__global__ __launch_bounds__(256)
void gaussian_rbf_wmma_kernel(const float* __restrict__ X,
                              const _Float16* __restrict__ Cpk,
                              const float* __restrict__ C2,
                              const float* __restrict__ IS2,
                              float* __restrict__ OUT) {
  const int lane = threadIdx.x;           // 0..31
  const int wave = threadIdx.y;           // 0..7
  const int hi   = (lane >= 16) ? 1 : 0;
  const int l16  = lane & 15;

  const int row_base = blockIdx.x * 16;
  const int col_base = wave * 64;
  const int tbase    = wave * 4;          // first of this wave's 4 column tiles

  const float* xrow = X + (size_t)(row_base + l16) * DIN;

  v8f acc0 = {0.f,0.f,0.f,0.f,0.f,0.f,0.f,0.f};
  v8f acc1 = acc0, acc2 = acc0, acc3 = acc0;
  float x2p = 0.f;

#pragma unroll 2
  for (int kbi = 0; kbi < NKB; ++kbi) {
    v16h a = cvt_frag_f16(xrow, kbi * 32, hi, x2p);
    const _Float16* bp = Cpk + (size_t)(((tbase * NKB) + kbi) * 32 + lane) * 16;
    v16h b0 = *(const v16h*)(bp + 0 * NKB * 32 * 16);
    v16h b1 = *(const v16h*)(bp + 1 * NKB * 32 * 16);
    v16h b2 = *(const v16h*)(bp + 2 * NKB * 32 * 16);
    v16h b3 = *(const v16h*)(bp + 3 * NKB * 32 * 16);
    acc0 = __builtin_amdgcn_wmma_f32_16x16x32_f16(false, a, false, b0, (short)0, acc0, false, false);
    acc1 = __builtin_amdgcn_wmma_f32_16x16x32_f16(false, a, false, b1, (short)0, acc1, false, false);
    acc2 = __builtin_amdgcn_wmma_f32_16x16x32_f16(false, a, false, b2, (short)0, acc2, false, false);
    acc3 = __builtin_amdgcn_wmma_f32_16x16x32_f16(false, a, false, b3, (short)0, acc3, false, false);
  }

  // Lane l and lane l^16 together cover the full K=128 row (l&15).
  float x2 = x2p + __shfl_xor(x2p, 16);
  // C/D layout: VGPR r -> (M = hi*8 + r, N = l16); x2[m] lives in lane m.
  float x2m[8];
#pragma unroll
  for (int r = 0; r < 8; ++r) x2m[r] = __shfl(x2, hi * 8 + r);

  const v8f accs[4] = {acc0, acc1, acc2, acc3};
#pragma unroll
  for (int t = 0; t < 4; ++t) {
    const int col = col_base + t * 16 + l16;
    const float c2     = C2[col];
    const float inv_s2 = IS2[col];
#pragma unroll
    for (int r = 0; r < 8; ++r) {
      const int m = hi * 8 + r;
      float d2 = fmaxf(fmaf(-2.0f, accs[t][r], x2m[r] + c2), 0.0f);
      OUT[(size_t)(row_base + m) * DOUTC + col] = __expf(-d2 * inv_s2);
    }
  }
}

extern "C" void kernel_launch(void* const* d_in, const int* in_sizes, int n_in,
                              void* d_out, int out_size, void* d_ws, size_t ws_size,
                              hipStream_t stream) {
  const float* X  = (const float*)d_in[0];   // [B, 128] fp32
  const float* C  = (const float*)d_in[1];   // [512, 128] fp32
  const float* LS = (const float*)d_in[2];   // [512] fp32
  float* OUT = (float*)d_out;                // [B, 512] fp32

  char* ws = (char*)d_ws;
  _Float16* Cpk = (_Float16*)ws;
  float* C2  = (float*)(ws + C2_OFF);
  float* IS2 = (float*)(ws + IS2_OFF);

  // Prologues: pack centers to f16 fragments; compute norms + inv sigma^2.
  pack_centers_kernel<<<dim3((NT * NKB * 32) / 256), dim3(256), 0, stream>>>(C, Cpk);
  center_norms_kernel<<<dim3((DOUTC + 255) / 256), dim3(256), 0, stream>>>(C, LS, C2, IS2);

  const int rows = in_sizes[0] / DIN;        // B = 131072 (multiple of 16)
  dim3 block(32, 8);                         // 8 wave32s
  dim3 grid(rows / 16);
  gaussian_rbf_wmma_kernel<<<grid, block, 0, stream>>>(X, Cpk, C2, IS2, OUT);
}
